// MaskedCausalAttention_48971217109725
// MI455X (gfx1250) — compile-verified, matrix operand order checked by reading, not run on hardware
//
#include <hip/hip_runtime.h>
#include <hip/hip_bf16.h>
#include <math.h>

// ---------------------------------------------------------------------------
// MaskedCausalAttention for MI455X (gfx1250): bf16 WMMA (fp32 accumulate)
//   1) qkv_kernel  : x@W.T+b for q/k/v, stores bf16 (q scaled by 1/sqrt(D),
//                    v stored TRANSPOSED (B,N,D,T) so PV B-frags are b128 loads)
//   2) attn_kernel : flash-attention per head, online softmax, 32-key steps
//   3) pool_kernel : group-mean + argmax -> cluster idx + per-head group lists
//   4) dot_kernel / loss_kernel : orthogonality loss -> d_out[0]
//   5) proj_kernel : cluster-GATHERED output projection (only the selected
//                    head per token: 16x fewer FLOPs than computing all heads)
// ---------------------------------------------------------------------------

typedef __attribute__((ext_vector_type(4)))  float  f32x4;
typedef __attribute__((ext_vector_type(8)))  float  v8f;
typedef __attribute__((ext_vector_type(4)))  __bf16 v4bf;
typedef __attribute__((ext_vector_type(8)))  __bf16 v8bf;
typedef __attribute__((ext_vector_type(16))) __bf16 v16bf;

#define WMMA_BF16(A, B, C) \
  __builtin_amdgcn_wmma_f32_16x16x32_bf16(false, (A), false, (B), (short)0, (C), false, false)

// 16-bit A/B fragment layout (ISA 7.12.2): lane%16 = M (or N); per lane the 16
// bf16 values are K = (r/4)*16 + (lane/16)*8 + (r%4)*2 + h, i.e. two contiguous
// 8-element (16-byte) runs at K-offsets {hi*8} and {16+hi*8}.
static __device__ inline v16bf join16(v8bf lo, v8bf hip) {
  v16bf r;
#pragma unroll
  for (int i = 0; i < 8; i++) { r[i] = lo[i]; r[8 + i] = hip[i]; }
  return r;
}

static __device__ inline v16bf frag16(const __bf16* base, int hi) {
  return join16(*(const v8bf*)(base + hi * 8), *(const v8bf*)(base + 16 + hi * 8));
}

// ---------------------------------------------------------------------------
// Kernel 1: fused QKV projection.  grid (BT/64, H/64, 3), block 128 (4 waves)
// ---------------------------------------------------------------------------
__global__ __launch_bounds__(128) void qkv_kernel(
    const float* __restrict__ x,
    const float* __restrict__ wq, const float* __restrict__ bq,
    const float* __restrict__ wk, const float* __restrict__ bk,
    const float* __restrict__ wv, const float* __restrict__ bv,
    __bf16* __restrict__ qb, __bf16* __restrict__ kb, __bf16* __restrict__ vb) {
  __shared__ __align__(16) __bf16 xs[64 * 32];
  __shared__ __align__(16) __bf16 wsm[64 * 32];
  const int z = blockIdx.z;
  const float* W  = (z == 0) ? wq : (z == 1 ? wk : wv);
  const float* Bs = (z == 0) ? bq : (z == 1 ? bk : bv);
  const int Rb = blockIdx.x * 64, Cb = blockIdx.y * 64;
  const int tid = threadIdx.x, wave = tid >> 5, lane = tid & 31;
  const int hi = lane >> 4, lo16 = lane & 15;
  v8f c[4] = {};
  for (int kbp = 0; kbp < 512; kbp += 32) {
#pragma unroll
    for (int j = 0; j < 4; j++) {               // stage 64x32 fp32 -> bf16 LDS
      int fl = (j * 128 + tid) * 4;
      int row = fl >> 5, col = fl & 31;
      f32x4 xv = *(const f32x4*)&x[(size_t)(Rb + row) * 512 + kbp + col];
      f32x4 wv4 = *(const f32x4*)&W[(size_t)(Cb + row) * 512 + kbp + col];
      v4bf xc, wc;
#pragma unroll
      for (int e = 0; e < 4; e++) { xc[e] = (__bf16)xv[e]; wc[e] = (__bf16)wv4[e]; }
      *(v4bf*)&xs[row * 32 + col] = xc;
      *(v4bf*)&wsm[row * 32 + col] = wc;
    }
    __syncthreads();
    int m = wave * 16 + lo16;
    v16bf a = frag16(&xs[m * 32], hi);
    v16bf bfr[4];                               // batch all B frags, then WMMAs
#pragma unroll
    for (int ns = 0; ns < 4; ns++) bfr[ns] = frag16(&wsm[(ns * 16 + lo16) * 32], hi);
#pragma unroll
    for (int ns = 0; ns < 4; ns++) c[ns] = WMMA_BF16(a, bfr[ns], c[ns]);
    __syncthreads();
  }
  const float qscale = 0.17677669529663687f;    // 1/sqrt(D=32) folded into q
#pragma unroll
  for (int ns = 0; ns < 4; ns++) {
    int j = Cb + ns * 16 + lo16;
    float bias = Bs[j];
    int hd = j >> 5, d = j & 31;
#pragma unroll
    for (int r = 0; r < 8; r++) {
      int i = Rb + wave * 16 + r + 8 * hi;      // C layout: M = r + 8*(lane/16)
      int bi = i >> 10, t = i & 1023;
      float v = c[ns][r] + bias;
      if (z == 0)      qb[((size_t)(bi * 16 + hd) * 1024 + t) * 32 + d] = (__bf16)(v * qscale);
      else if (z == 1) kb[((size_t)(bi * 16 + hd) * 1024 + t) * 32 + d] = (__bf16)v;
      else             vb[((size_t)(bi * 16 + hd) * 32 + d) * 1024 + t] = (__bf16)v; // V^T
    }
  }
}

// ---------------------------------------------------------------------------
// Kernel 2: causal flash attention. grid (T/64, B*N), block 128 (4 waves).
// Each wave owns a 16-row query tile; 32-key steps => 4 WMMAs per step.
// V B-frag loads are issued BEFORE the softmax VALU chain so global latency
// hides under exp/shuffle work; next K/V tiles are prefetched.
// ---------------------------------------------------------------------------
__global__ __launch_bounds__(128) void attn_kernel(
    const __bf16* __restrict__ qb, const __bf16* __restrict__ kbuf,
    const __bf16* __restrict__ vb, float* __restrict__ att) {
  __shared__ __align__(16) __bf16 pbuf[4][16 * 32];   // per-wave P transpose
  const int hb = blockIdx.y;                          // b*16 + head
  const int tid = threadIdx.x, wave = tid >> 5, lane = tid & 31;
  const int hi = lane >> 4, lo16 = lane & 15;
  const __bf16* qh = qb + (size_t)hb * 32768;
  const __bf16* kh = kbuf + (size_t)hb * 32768;
  const __bf16* vh = vb + (size_t)hb * 32768;
  const int Rw = blockIdx.x * 64 + wave * 16;
  const int m = Rw + lo16;
  v16bf aq = frag16(&qh[m * 32], hi);
  v8f o0 = {}, o1 = {};
  float mrow[8], lrow[8];
#pragma unroll
  for (int r = 0; r < 8; r++) { mrow[r] = -__builtin_inff(); lrow[r] = 0.f; }
  __bf16* pb = &pbuf[wave][0];
  for (int jb = 0; jb <= Rw + 15; jb += 32) {
    int key0 = jb + lo16, key1 = jb + 16 + lo16;
    v16bf bk0 = frag16(&kh[key0 * 32], hi);
    v16bf bk1 = frag16(&kh[key1 * 32], hi);
    v8f zc = {};
    v8f s0 = WMMA_BF16(aq, bk0, zc);
    v8f s1 = WMMA_BF16(aq, bk1, zc);
    // V fragments: independent of softmax -> issue early (latency hiding)
    v16bf bv0 = frag16(&vh[lo16 * 1024 + jb], hi);        // V^T: K contiguous
    v16bf bv1 = frag16(&vh[(16 + lo16) * 1024 + jb], hi);
    if (jb + 32 <= Rw + 15) {                   // prefetch next tiles into L2
      __builtin_prefetch(&kh[(jb + 32 + lo16) * 32], 0, 3);
      __builtin_prefetch(&vh[lo16 * 1024 + jb + 32], 0, 3);
    }
#pragma unroll
    for (int r = 0; r < 8; r++) {
      int row = Rw + r + 8 * hi;
      float e0 = (key0 <= row) ? s0[r] : -__builtin_inff();
      float e1 = (key1 <= row) ? s1[r] : -__builtin_inff();
      float t = fmaxf(e0, e1);                  // row max over 16 lanes (half)
      t = fmaxf(t, __shfl_xor(t, 1, 32));
      t = fmaxf(t, __shfl_xor(t, 2, 32));
      t = fmaxf(t, __shfl_xor(t, 4, 32));
      t = fmaxf(t, __shfl_xor(t, 8, 32));
      float mn = fmaxf(mrow[r], t);
      float alpha = __expf(mrow[r] - mn);
      float p0 = __expf(e0 - mn);
      float p1 = __expf(e1 - mn);
      float rs = p0 + p1;
      rs += __shfl_xor(rs, 1, 32);
      rs += __shfl_xor(rs, 2, 32);
      rs += __shfl_xor(rs, 4, 32);
      rs += __shfl_xor(rs, 8, 32);
      lrow[r] = lrow[r] * alpha + rs;
      o0[r] = o0[r] * alpha;
      o1[r] = o1[r] * alpha;
      mrow[r] = mn;
      pb[(r + 8 * hi) * 32 + lo16] = (__bf16)p0;        // C-layout -> A-layout
      pb[(r + 8 * hi) * 32 + 16 + lo16] = (__bf16)p1;   // transpose via LDS
    }
    v16bf ap = frag16(&pb[lo16 * 32], hi);
    o0 = WMMA_BF16(ap, bv0, o0);
    o1 = WMMA_BF16(ap, bv1, o1);
  }
  float* ah = att + (size_t)hb * 32768;
#pragma unroll
  for (int r = 0; r < 8; r++) {
    float inv = 1.0f / lrow[r];
    int row = Rw + r + 8 * hi;
    ah[row * 32 + lo16] = o0[r] * inv;
    ah[row * 32 + 16 + lo16] = o1[r] * inv;
  }
}

// ---------------------------------------------------------------------------
// Kernel 3: avgpool + argmax -> cluster idx (d_out tail) + per-head lists.
// ---------------------------------------------------------------------------
__global__ __launch_bounds__(256) void pool_kernel(
    const float* __restrict__ att, float* __restrict__ dout,
    int* __restrict__ counts, int* __restrict__ glist) {
  int b = blockIdx.x, g = threadIdx.x;
  float best = -__builtin_inff();
  int bi = 0;
  for (int n = 0; n < 16; n++) {
    const float* p = att + ((size_t)(b * 16 + n) * 1024 + g * 4) * 32;
    float s = 0.f;
    for (int i = 0; i < 128; i++) s += p[i];
    if (s > best) { best = s; bi = n; }          // strict > == first-argmax
  }
  dout[1 + 4194304 + b * 256 + g] = (float)bi;
  int slot = atomicAdd(&counts[b * 16 + bi], 1);
  glist[(b * 16 + bi) * 256 + slot] = g;
}

// ---------------------------------------------------------------------------
// Kernel 4: pairwise head dots (incl. diagonal = squared norms).
// ---------------------------------------------------------------------------
__global__ __launch_bounds__(256) void dot_kernel(
    const float* __restrict__ att, float* __restrict__ dots) {
  __shared__ float red[256];
  int blk = blockIdx.x;
  int b = blk / 136, pp = blk % 136;
  int n = 0, mm = 0, acc = 0;
  for (int i = 0; i < 16; i++) {
    int cc = 16 - i;
    if (pp < acc + cc) { n = i; mm = i + (pp - acc); break; }
    acc += cc;
  }
  const float* fa = att + (size_t)(b * 16 + n) * 32768;
  const float* fb = att + (size_t)(b * 16 + mm) * 32768;
  float s = 0.f;
  for (int k = threadIdx.x; k < 32768; k += 256) s += fa[k] * fb[k];
  red[threadIdx.x] = s;
  __syncthreads();
  for (int st = 128; st > 0; st >>= 1) {
    if (threadIdx.x < st) red[threadIdx.x] += red[threadIdx.x + st];
    __syncthreads();
  }
  if (threadIdx.x == 0) {
    dots[(b * 16 + n) * 16 + mm] = red[0];
    dots[(b * 16 + mm) * 16 + n] = red[0];
  }
}

// ---------------------------------------------------------------------------
// Kernel 5: orthogonality loss -> d_out[0].
// ---------------------------------------------------------------------------
__global__ __launch_bounds__(64) void loss_kernel(
    const float* __restrict__ dots, float* __restrict__ dout) {
  __shared__ float lb[8];
  int t = threadIdx.x;
  if (t < 8) {
    const float* d = dots + t * 256;
    float nrm[16];
    for (int n = 0; n < 16; n++) nrm[n] = fmaxf(sqrtf(d[n * 16 + n]), 1e-12f);
    float s = 0.f;
    for (int n = 0; n < 16; n++)
      for (int mm = 0; mm < 16; mm++) {
        float v = d[n * 16 + mm] / (nrm[n] * nrm[mm]);
        float e = v - ((n == mm) ? 1.f : 0.f);
        float e2 = e * e;
        s += e2 * e2;                            // (mult-eye)^4
      }
    lb[t] = sqrtf(s);
  }
  __syncthreads();
  if (t == 0) {
    float s = 0.f;
    for (int i = 0; i < 8; i++) s += lb[i];
    dout[0] = s / 8.f;
  }
}

// ---------------------------------------------------------------------------
// Kernel 6: cluster-gathered output projection (only the selected head per
// token => 16x fewer FLOPs). grid (16 chunks, 16 heads, 8 batches), 256 thr.
// A-frag global loads are issued BEFORE the LDS staging of wp[n] so they are
// in flight during staging; B-frags batched 4 at a time ahead of the WMMAs.
// ---------------------------------------------------------------------------
__global__ __launch_bounds__(256) void proj_kernel(
    const float* __restrict__ att, const float* __restrict__ wp,
    const float* __restrict__ bp, const int* __restrict__ counts,
    const int* __restrict__ glist, float* __restrict__ dout) {
  __shared__ __align__(16) __bf16 wls[512 * 32];
  const int b = blockIdx.z, n = blockIdx.y, g0 = blockIdx.x * 16;
  const int cnt = counts[b * 16 + n];
  if (g0 >= cnt) return;                         // block-uniform exit
  const int tid = threadIdx.x, wave = tid >> 5, lane = tid & 31;
  const int hi = lane >> 4, lo16 = lane & 15;
  const int wsub = wave & 3, nhalf = wave >> 2;
  const int* gl = glist + (b * 16 + n) * 256;
  int mrowA = wsub * 16 + lo16;
  int grpA = g0 + (mrowA >> 2);
  int tokA = ((grpA < cnt) ? gl[grpA] : 0) * 4 + (mrowA & 3);
  const float* wpn = wp + (size_t)n * 262144;
  v8f c[16] = {};
  for (int kb2 = 0; kb2 < 512; kb2 += 32) {
    int hh = kb2 >> 5;
    // A fragment loads first: independent of LDS, overlap with staging below
    const float* arow = att + ((size_t)(b * 16 + hh) * 1024 + tokA) * 32;
    f32x4 a0 = *(const f32x4*)&arow[hi * 8];
    f32x4 a1 = *(const f32x4*)&arow[hi * 8 + 4];
    f32x4 a2 = *(const f32x4*)&arow[16 + hi * 8];
    f32x4 a3 = *(const f32x4*)&arow[16 + hi * 8 + 4];
#pragma unroll
    for (int j = 0; j < 16; j++) {               // stage wp[n][:,kb:kb+32]
      int fl = (j * 256 + tid) * 4;
      int row = fl >> 5, col = fl & 31;
      f32x4 wv4 = *(const f32x4*)&wpn[(size_t)row * 512 + kb2 + col];
      v4bf wc;
#pragma unroll
      for (int e = 0; e < 4; e++) wc[e] = (__bf16)wv4[e];
      *(v4bf*)&wls[row * 32 + col] = wc;
    }
    v16bf a;
#pragma unroll
    for (int e = 0; e < 4; e++) {
      a[e] = (__bf16)a0[e]; a[4 + e] = (__bf16)a1[e];
      a[8 + e] = (__bf16)a2[e]; a[12 + e] = (__bf16)a3[e];
    }
    __syncthreads();
#pragma unroll
    for (int grp4 = 0; grp4 < 4; grp4++) {       // 4 frags batched, 4 WMMAs
      v16bf bfr[4];
#pragma unroll
      for (int q = 0; q < 4; q++) {
        int jrow = nhalf * 256 + (grp4 * 4 + q) * 16 + lo16;
        bfr[q] = frag16(&wls[jrow * 32], hi);
      }
#pragma unroll
      for (int q = 0; q < 4; q++)
        c[grp4 * 4 + q] = WMMA_BF16(a, bfr[q], c[grp4 * 4 + q]);
    }
    __syncthreads();
  }
  int tokR[8], okR[8];
#pragma unroll
  for (int r = 0; r < 8; r++) {
    int mr = wsub * 16 + r + 8 * hi;
    int grp = g0 + (mr >> 2);
    okR[r] = (grp < cnt);
    tokR[r] = (okR[r] ? gl[grp] : 0) * 4 + (mr & 3);
  }
#pragma unroll
  for (int ns = 0; ns < 16; ns++) {
    int jcol = nhalf * 256 + ns * 16 + lo16;
    float bias = bp[n * 512 + jcol];
#pragma unroll
    for (int r = 0; r < 8; r++) {
      if (okR[r]) dout[1 + ((size_t)(b * 1024 + tokR[r])) * 512 + jcol] = c[ns][r] + bias;
    }
  }
}

// ---------------------------------------------------------------------------
// Launch. ws layout (bytes):
//   [0, 8M)   q bf16 (B,N,T,D)        [8M, 16M)  k bf16 (B,N,T,D)
//   [16M,24M) v bf16 (B,N,D,T)        [24M,40M)  att f32 (B,N,T,D)
//   [40M, +8K) dots  [+8K, +512) counts  [+16K, +128K) glist
// ---------------------------------------------------------------------------
extern "C" void kernel_launch(void* const* d_in, const int* in_sizes, int n_in,
                              void* d_out, int out_size, void* d_ws, size_t ws_size,
                              hipStream_t stream) {
  (void)in_sizes; (void)n_in; (void)out_size; (void)ws_size;
  const float* x  = (const float*)d_in[0];
  const float* wq = (const float*)d_in[1];
  const float* bq = (const float*)d_in[2];
  const float* wk = (const float*)d_in[3];
  const float* bk = (const float*)d_in[4];
  const float* wv = (const float*)d_in[5];
  const float* bv = (const float*)d_in[6];
  const float* wp = (const float*)d_in[7];
  const float* bp = (const float*)d_in[8];
  float* out = (float*)d_out;
  char* ws = (char*)d_ws;
  __bf16* qb   = (__bf16*)(ws + 0);
  __bf16* kb   = (__bf16*)(ws + ((size_t)8 << 20));
  __bf16* vb   = (__bf16*)(ws + ((size_t)16 << 20));
  float*  att  = (float*)(ws + ((size_t)24 << 20));
  float*  dots = (float*)(ws + ((size_t)40 << 20));
  int* counts  = (int*)(ws + ((size_t)40 << 20) + 8192);
  int* glist   = (int*)(ws + ((size_t)40 << 20) + 16384);

  hipMemsetAsync(counts, 0, 1024, stream);
  qkv_kernel<<<dim3(128, 8, 3), 128, 0, stream>>>(x, wq, bq, wk, bk, wv, bv, qb, kb, vb);
  attn_kernel<<<dim3(16, 128), 128, 0, stream>>>(qb, kb, vb, att);
  pool_kernel<<<dim3(8), 256, 0, stream>>>(att, out, counts, glist);
  dot_kernel<<<dim3(1088), 256, 0, stream>>>(att, dots);
  loss_kernel<<<dim3(1), 64, 0, stream>>>(dots, out);
  proj_kernel<<<dim3(16, 16, 8), 256, 0, stream>>>(att, wp, bp, counts, glist, out);
}
